// RBFGaussianLayer_24137716204066
// MI455X (gfx1250) — matrix operand order repaired and checked by reading, not compiled, not run on hardware
//
#include <hip/hip_runtime.h>

// ---------------------------------------------------------------------------
// RBF Gaussian layer for MI455X (gfx1250, wave32, WMMA).
//   out[b,m] = exp(-beta[m] * (||x_b||^2 + ||c_m||^2 - 2 * x_b . c_m))
// GEMM via V_WMMA_F32_16X16X32_BF16 with bf16 hi/lo split-precision (3 WMMAs
// per fp32 MAC tile: hi*hi + hi*lo + lo*hi) for near-fp32 accuracy at bf16
// matrix-core rates.
// ---------------------------------------------------------------------------

typedef __bf16 bf16_t;
typedef __attribute__((ext_vector_type(8)))  __bf16 v8bf;
typedef __attribute__((ext_vector_type(16))) __bf16 v16bf;
typedef __attribute__((ext_vector_type(8)))  float  v8f;

#define IN_F   512
#define NCTR   2048
#define BATCH  16384

#define BM   128          // block tile M (batch rows)
#define BN   128          // block tile N (centers)
#define BK   32           // k-tile == WMMA K for bf16
#define LDSK 40           // padded LDS row stride (bf16 elems, 80B) -> bank spread
#define NT   (IN_F / BK)  // 16 k-tiles

__device__ __forceinline__ v16bf cat8(v8bf a, v8bf b) {
  return __builtin_shufflevector(a, b, 0,1,2,3,4,5,6,7,8,9,10,11,12,13,14,15);
}

// split 16 fp32 into bf16 hi/lo and store packed (two v8bf each) to LDS
__device__ __forceinline__ void cvt_store16(const float* f, bf16_t* dhi, bf16_t* dlo) {
  v8bf h0, h1, l0, l1;
#pragma unroll
  for (int j = 0; j < 8; ++j) {
    bf16_t h = (bf16_t)f[j];
    h0[j] = h;
    l0[j] = (bf16_t)(f[j] - (float)h);
  }
#pragma unroll
  for (int j = 0; j < 8; ++j) {
    bf16_t h = (bf16_t)f[8 + j];
    h1[j] = h;
    l1[j] = (bf16_t)(f[8 + j] - (float)h);
  }
  *(v8bf*)dhi       = h0;
  *(v8bf*)(dhi + 8) = h1;
  *(v8bf*)dlo       = l0;
  *(v8bf*)(dlo + 8) = l1;
}

// ---------------------------------------------------------------------------
// Kernel A: row squared-norms. One wave32 per row (512 floats = 4 float4/lane).
// ---------------------------------------------------------------------------
__global__ __launch_bounds__(256) void rbf_rownorm(const float* __restrict__ src,
                                                   float* __restrict__ dst,
                                                   int nrows) {
  const int wave = (int)((blockIdx.x * blockDim.x + threadIdx.x) >> 5);
  const int lane = (int)(threadIdx.x & 31);
  if (wave >= nrows) return;
  const float4* p = (const float4*)(src + (size_t)wave * IN_F);
  float s = 0.0f;
#pragma unroll
  for (int i = 0; i < 4; ++i) {
    float4 v = p[lane + 32 * i];
    s += v.x * v.x + v.y * v.y + v.z * v.z + v.w * v.w;
  }
#pragma unroll
  for (int off = 16; off > 0; off >>= 1) s += __shfl_xor(s, off, 32);
  if (lane == 0) dst[wave] = s;
}

// ---------------------------------------------------------------------------
// Kernel B: 128x128 tile GEMM (x @ centers^T) + fused RBF epilogue.
// 256 threads = 8 waves in a 2(M) x 4(N) grid; each wave owns 4x2 WMMA tiles.
// ---------------------------------------------------------------------------
__global__ __launch_bounds__(256) void rbf_gemm(const float* __restrict__ x,
                                                const float* __restrict__ centers,
                                                const float* __restrict__ beta,
                                                const float* __restrict__ xsq,
                                                const float* __restrict__ csq,
                                                float* __restrict__ out) {
  extern __shared__ __align__(16) unsigned char smraw[];
  bf16_t* a_hi = (bf16_t*)smraw;                  // [2][BM][LDSK]
  bf16_t* a_lo = a_hi + 2 * BM * LDSK;
  bf16_t* b_hi = a_lo + 2 * BM * LDSK;
  bf16_t* b_lo = b_hi + 2 * BN * LDSK;

  const int t      = (int)threadIdx.x;
  const int lane   = t & 31;
  const int waveId = t >> 5;       // 0..7
  const int waveM  = waveId >> 2;  // 0..1  -> 64 M-rows per wave
  const int waveN  = waveId & 3;   // 0..3  -> 32 N-cols per wave
  const int lrow   = lane & 15;    // M (A frag) / N (B frag) index
  const int lhalf  = lane >> 4;    // K-half selector

  const int blockM = (int)blockIdx.y * BM;
  const int blockN = (int)blockIdx.x * BN;

  // global staging: each thread owns 16 consecutive floats of one tile row
  const int ldRow = t >> 1;
  const int ldCol = (t & 1) * 16;
  const float* aSrc = x       + (size_t)(blockM + ldRow) * IN_F + ldCol;
  const float* bSrc = centers + (size_t)(blockN + ldRow) * IN_F + ldCol;

  v8f acc[4][2];
#pragma unroll
  for (int i = 0; i < 4; ++i)
#pragma unroll
    for (int j = 0; j < 2; ++j)
#pragma unroll
      for (int r = 0; r < 8; ++r) acc[i][j][r] = 0.0f;

  float fa[16], fb[16];

  // preload k-tile 0 into LDS stage 0
  {
    const float4* pa = (const float4*)aSrc;
    const float4* pb = (const float4*)bSrc;
#pragma unroll
    for (int i = 0; i < 4; ++i) {
      *(float4*)&fa[4 * i] = pa[i];
      *(float4*)&fb[4 * i] = pb[i];
    }
    cvt_store16(fa, &a_hi[ldRow * LDSK + ldCol], &a_lo[ldRow * LDSK + ldCol]);
    cvt_store16(fb, &b_hi[ldRow * LDSK + ldCol], &b_lo[ldRow * LDSK + ldCol]);
  }
  __syncthreads();

  for (int kt = 0; kt < NT; ++kt) {
    const int cur = kt & 1;
    const int nxt = cur ^ 1;

    // issue next global tile early; compute below hides the latency
    if (kt + 1 < NT) {
      const float4* pa = (const float4*)(aSrc + (kt + 1) * BK);
      const float4* pb = (const float4*)(bSrc + (kt + 1) * BK);
#pragma unroll
      for (int i = 0; i < 4; ++i) {
        *(float4*)&fa[4 * i] = pa[i];
        *(float4*)&fb[4 * i] = pb[i];
      }
    }

    const bf16_t* aH = a_hi + cur * BM * LDSK;
    const bf16_t* aL = a_lo + cur * BM * LDSK;
    const bf16_t* bH = b_hi + cur * BN * LDSK;
    const bf16_t* bL = b_lo + cur * BN * LDSK;

    // B fragments: lane = column n (== centers row), contiguous K = lhalf*16..
    v16bf bh[2], bl[2];
#pragma unroll
    for (int tn = 0; tn < 2; ++tn) {
      const int br = waveN * 32 + tn * 16 + lrow;
      const bf16_t* ph = bH + br * LDSK + lhalf * 16;
      const bf16_t* pl = bL + br * LDSK + lhalf * 16;
      bh[tn] = cat8(*(const v8bf*)ph, *(const v8bf*)(ph + 8));
      bl[tn] = cat8(*(const v8bf*)pl, *(const v8bf*)(pl + 8));
    }

#pragma unroll
    for (int tm = 0; tm < 4; ++tm) {
      const int ar = waveM * 64 + tm * 16 + lrow;
      const bf16_t* ph = aH + ar * LDSK;
      const bf16_t* pl = aL + ar * LDSK;
      // A fragment: contiguous K = lhalf*16 .. +15
      v16bf ahf = cat8(*(const v8bf*)(ph + lhalf * 16),
                       *(const v8bf*)(ph + lhalf * 16 + 8));
      v16bf alf = cat8(*(const v8bf*)(pl + lhalf * 16),
                       *(const v8bf*)(pl + lhalf * 16 + 8));
#pragma unroll
      for (int tn = 0; tn < 2; ++tn) {
        acc[tm][tn] = __builtin_amdgcn_wmma_f32_16x16x32_bf16(
            false, ahf, false, bh[tn], (short)0, acc[tm][tn], false, false);
        acc[tm][tn] = __builtin_amdgcn_wmma_f32_16x16x32_bf16(
            false, ahf, false, bl[tn], (short)0, acc[tm][tn], false, false);
        acc[tm][tn] = __builtin_amdgcn_wmma_f32_16x16x32_bf16(
            false, alf, false, bh[tn], (short)0, acc[tm][tn], false, false);
      }
    }

    if (kt + 1 < NT) {
      cvt_store16(fa, &a_hi[nxt * BM * LDSK + ldRow * LDSK + ldCol],
                      &a_lo[nxt * BM * LDSK + ldRow * LDSK + ldCol]);
      cvt_store16(fb, &b_hi[nxt * BN * LDSK + ldRow * LDSK + ldCol],
                      &b_lo[nxt * BN * LDSK + ldRow * LDSK + ldCol]);
    }
    __syncthreads();
  }

  // ---- epilogue: exp(-beta * (xsq + csq - 2*cross)), norms staged via LDS ----
  float* fs = (float*)smraw;
  float* xs = fs;            // [BM]
  float* cs = fs + BM;       // [BN]
  float* bs = fs + BM + BN;  // [BN]
  if (t < BM) {
    xs[t] = xsq[blockM + t];
  } else {
    cs[t - BM] = csq[blockN + (t - BM)];
    bs[t - BM] = beta[blockN + (t - BM)];
  }
  __syncthreads();

#pragma unroll
  for (int tm = 0; tm < 4; ++tm) {
#pragma unroll
    for (int tn = 0; tn < 2; ++tn) {
#pragma unroll
      for (int r = 0; r < 8; ++r) {
        // C/D layout: VGPR r, lane L -> M = r + 8*(L>=16), N = L&15
        const int m = waveM * 64 + tm * 16 + lhalf * 8 + r;
        const int n = waveN * 32 + tn * 16 + lrow;
        const float d = xs[m] + cs[n] - 2.0f * acc[tm][tn][r];
        out[(size_t)(blockM + m) * NCTR + (blockN + n)] = __expf(-bs[n] * d);
      }
    }
  }
}

// ---------------------------------------------------------------------------
extern "C" void kernel_launch(void* const* d_in, const int* in_sizes, int n_in,
                              void* d_out, int out_size, void* d_ws, size_t ws_size,
                              hipStream_t stream) {
  const float* x       = (const float*)d_in[0];
  const float* centers = (const float*)d_in[1];
  const float* beta    = (const float*)d_in[2];
  float*       out     = (float*)d_out;

  float* xsq = (float*)d_ws;       // BATCH floats
  float* csq = xsq + BATCH;        // NCTR floats

  const size_t smem = (size_t)(2 * BM * LDSK + 2 * BM * LDSK +
                               2 * BN * LDSK + 2 * BN * LDSK) * sizeof(bf16_t); // 80 KB
  hipFuncSetAttribute((const void*)rbf_gemm,
                      hipFuncAttributeMaxDynamicSharedMemorySize, (int)smem);

  // row norms: 8 waves per 256-thread block -> 8 rows per block
  rbf_rownorm<<<BATCH / 8, 256, 0, stream>>>(x, xsq, BATCH);
  rbf_rownorm<<<NCTR / 8, 256, 0, stream>>>(centers, csq, NCTR);

  dim3 grid(NCTR / BN, BATCH / BM);  // (16, 128)
  rbf_gemm<<<grid, 256, smem, stream>>>(x, centers, beta, xsq, csq, out);
}
